// MultiHeadAttention_33165737460163
// MI455X (gfx1250) — matrix operand order repaired and checked by reading, not compiled, not run on hardware
//
#include <hip/hip_runtime.h>

// ---------------------------------------------------------------------------
// CDNA5 (gfx1250) multi-head attention: bf16 WMMA (v_wmma_f32_16x16x32_bf16),
// fp32 accumulate, flash-attention softmax, wave32, async global->LDS staging.
// ---------------------------------------------------------------------------

typedef __attribute__((ext_vector_type(16))) __bf16 v16bf;
typedef __attribute__((ext_vector_type(8)))  float  v8f;

union FragBF { v16bf v; unsigned u[8]; };
union Pack8  { uint4 q; unsigned short h[8]; };

// async 16B global->LDS copy (CDNA5 TDM-adjacent path, ASYNCcnt-tracked)
#define ASYNC_LDS_B128(ldsoff, gptr, OFF)                                   \
  asm volatile("global_load_async_to_lds_b128 %0, %1, off offset:" OFF      \
               : : "v"(ldsoff), "v"(gptr) : "memory")

__device__ __forceinline__ unsigned short f2bf(float f) {
  unsigned u = __float_as_uint(f);
  u = (u + 0x7FFFu + ((u >> 16) & 1u)) >> 16;   // round-to-nearest-even
  return (unsigned short)u;
}

__device__ __forceinline__ float rowmax16(float x) {
#pragma unroll
  for (int m = 1; m < 16; m <<= 1) x = fmaxf(x, __shfl_xor(x, m, 32));
  return x;
}
__device__ __forceinline__ float rowsum16(float x) {
#pragma unroll
  for (int m = 1; m < 16; m <<= 1) x += __shfl_xor(x, m, 32);
  return x;
}

// ---------------- fp32 -> bf16 bulk convert (4 elems / thread) -------------
__global__ __launch_bounds__(256)
void cvt_bf16_kernel(const float* __restrict__ src, unsigned short* __restrict__ dst,
                     long long n4) {
  long long i = (long long)blockIdx.x * blockDim.x + threadIdx.x;
  if (i >= n4) return;
  float4 f = ((const float4*)src)[i];
  uint2 o;
  o.x = (unsigned)f2bf(f.x) | ((unsigned)f2bf(f.y) << 16);
  o.y = (unsigned)f2bf(f.z) | ((unsigned)f2bf(f.w) << 16);
  ((uint2*)dst)[i] = o;
}

// ---------------- generic bf16 WMMA GEMM, BM=128 BN=128 BK=32 --------------
// 8 waves arranged 4(M) x 2(N): each wave computes 32 rows x 64 cols.
// Register-staged pipeline + global_prefetch_b8 two tiles ahead.

struct QKVEpi {
  const float* bias;             // [6144]
  unsigned short* q;             // [B,H,T,D]
  unsigned short* k;             // [B,H,T,D]
  unsigned short* vT;            // [B,H,D,T]
  __device__ __forceinline__ void store(int m, int n, float val) const {
    val += bias[n];
    const int which = n >> 11;          // 0=q 1=k 2=v   (C = 2048)
    const int h = (n >> 7) & 15;        // D = 128
    const int d = n & 127;
    const int b = m >> 11;              // T = 2048
    const int t = m & 2047;
    const unsigned short bv = f2bf(val);
    const size_t bh = (size_t)(b * 16 + h);
    if (which == 0)      q [(bh * 2048 + t) * 128 + d] = bv;
    else if (which == 1) k [(bh * 2048 + t) * 128 + d] = bv;
    else                 vT[(bh * 128  + d) * 2048 + t] = bv;
  }
};

struct ProjEpi {
  const float* bias;             // [N]
  float* out;                    // [M x N]
  int N;
  __device__ __forceinline__ void store(int m, int n, float val) const {
    out[(size_t)m * N + n] = val + bias[n];
  }
};

template <class Epi>
__global__ __launch_bounds__(256)
void gemm_bf16(const unsigned short* __restrict__ A, const unsigned short* __restrict__ B,
               int M, int N, int K, Epi epi) {
  __shared__ unsigned short As[128 * 36];   // [row][k] pad->36 halves (72B rows)
  __shared__ unsigned short Bs[128 * 36];   // k-transposed: [n][k]

  const int tid  = threadIdx.x;
  const int lane = tid & 31;
  const int wave = tid >> 5;
  const int lo   = lane & 15;
  const int hi   = lane >> 4;
  const int bm   = blockIdx.y * 128;
  const int bn   = blockIdx.x * 128;
  const int mslab = (wave & 3) * 32;        // wave's 32-row slab
  const int nhalf = (wave >> 2) * 64;       // wave's 64-col half

  const int ar  = tid >> 1;                 // A row 0..127
  const int ac  = (tid & 1) * 16;           // A col offset (halves)
  const int br  = tid >> 4;                 // B row 0..15 (and +16)
  const int bc8 = (tid & 15) * 8;           // B col offset (halves)

  const unsigned short* aSrc  = A + (size_t)(bm + ar) * K + ac;
  const unsigned short* bSrc0 = B + (size_t)br        * N + bn + bc8;
  const unsigned short* bSrc1 = B + (size_t)(br + 16) * N + bn + bc8;

  v8f acc[2][4];
#pragma unroll
  for (int i = 0; i < 2; ++i)
#pragma unroll
    for (int s = 0; s < 4; ++s)
#pragma unroll
      for (int r = 0; r < 8; ++r) acc[i][s][r] = 0.f;

  // prime: global loads for tile 0
  uint4 aReg0 = ((const uint4*)(aSrc))[0];
  uint4 aReg1 = ((const uint4*)(aSrc))[1];
  Pack8 bReg0, bReg1;
  bReg0.q = ((const uint4*)(bSrc0))[0];
  bReg1.q = ((const uint4*)(bSrc1))[0];

  for (int k0 = 0; k0 < K; k0 += 32) {
    // ---- stage current registers into LDS ----
    {
      uint2* dst = (uint2*)&As[ar * 36 + ac];
      dst[0] = make_uint2(aReg0.x, aReg0.y);
      dst[1] = make_uint2(aReg0.z, aReg0.w);
      dst[2] = make_uint2(aReg1.x, aReg1.y);
      dst[3] = make_uint2(aReg1.z, aReg1.w);
#pragma unroll
      for (int i = 0; i < 8; ++i) {
        Bs[(bc8 + i) * 36 + br]      = bReg0.h[i];
        Bs[(bc8 + i) * 36 + br + 16] = bReg1.h[i];
      }
    }
    __syncthreads();

    // ---- issue global loads for tile k0+32 (overlaps compute below) ----
    if (k0 + 32 < K) {
      const unsigned short* aN = aSrc + (k0 + 32);
      const unsigned short* b0 = bSrc0 + (size_t)(k0 + 32) * N;
      const unsigned short* b1 = bSrc1 + (size_t)(k0 + 32) * N;
      aReg0 = ((const uint4*)aN)[0];
      aReg1 = ((const uint4*)aN)[1];
      bReg0.q = ((const uint4*)b0)[0];
      bReg1.q = ((const uint4*)b1)[0];
      if (k0 + 64 < K) {
        __builtin_prefetch(aSrc + (k0 + 64), 0, 1);
        __builtin_prefetch(bSrc0 + (size_t)(k0 + 64) * N, 0, 1);
        __builtin_prefetch(bSrc1 + (size_t)(k0 + 64) * N, 0, 1);
      }
    }

    // ---- compute: 2 A-frags x 4 B-frags -> 8 WMMAs ----
    FragBF af[2];
#pragma unroll
    for (int i = 0; i < 2; ++i) {
      const unsigned* As32 = (const unsigned*)&As[(mslab + i * 16 + lo) * 36];
#pragma unroll
      for (int v = 0; v < 8; ++v) {
        const int kk = (v & 3) * 2 + (v >> 2) * 16 + hi * 8;
        af[i].u[v] = As32[kk >> 1];
      }
    }
#pragma unroll
    for (int s = 0; s < 4; ++s) {
      FragBF bf_;
      const unsigned* Bs32 = (const unsigned*)&Bs[(nhalf + s * 16 + lo) * 36];
#pragma unroll
      for (int v = 0; v < 8; ++v) {
        const int kk = (v & 3) * 2 + (v >> 2) * 16 + hi * 8;
        bf_.u[v] = Bs32[kk >> 1];
      }
      acc[0][s] = __builtin_amdgcn_wmma_f32_16x16x32_bf16(false, af[0].v, false, bf_.v,
                                                          (short)0, acc[0][s], false, false);
      acc[1][s] = __builtin_amdgcn_wmma_f32_16x16x32_bf16(false, af[1].v, false, bf_.v,
                                                          (short)0, acc[1][s], false, false);
    }
    __syncthreads();
  }

#pragma unroll
  for (int i = 0; i < 2; ++i)
#pragma unroll
    for (int s = 0; s < 4; ++s)
#pragma unroll
      for (int r = 0; r < 8; ++r)
        epi.store(bm + mslab + i * 16 + r + hi * 8,
                  bn + nhalf + s * 16 + lo, acc[i][s][r]);
}

// ---------------- flash attention: 64 q rows / block, 4 waves --------------
// Q,K: [B,H,T,D] bf16.  VT: [B,H,D,T] bf16.  Y: [B,T,C] bf16 output.
// K / V^T tiles staged with global_load_async_to_lds_b128 (ASYNCcnt).
__global__ __launch_bounds__(128)
void attn_kernel(const unsigned short* __restrict__ Q, const unsigned short* __restrict__ Kb,
                 const unsigned short* __restrict__ VT, unsigned short* __restrict__ Y) {
  const int T = 2048, D = 128, C = 2048;
  __shared__ unsigned short Ks[64 * 136];      // [key][d]   272B rows (16B mult)
  __shared__ unsigned short Vs[128 * 72];      // [d][key]   144B rows (16B mult)
  __shared__ unsigned short Ps[4 * 16 * 68];   // per-wave P tile [row][key]

  const int tid  = threadIdx.x;
  const int lane = tid & 31;
  const int wave = tid >> 5;
  const int lo   = lane & 15;
  const int hi   = lane >> 4;
  const int qt   = blockIdx.x;                 // query tile (64 rows)
  const int bh   = blockIdx.y;                 // b*16 + h
  const int b    = bh >> 4, h = bh & 15;
  const int qb   = qt * 64;
  const float scale = 0.0883883476483184f;     // 1/sqrt(128)

  // Q fragments resident in registers (4 K-steps of 32 over D=128)
  FragBF qf[4];
  {
    const unsigned* qrow =
        (const unsigned*)(Q + ((size_t)bh * T + qb + wave * 16 + lo) * D);
#pragma unroll
    for (int ks = 0; ks < 4; ++ks)
#pragma unroll
      for (int v = 0; v < 8; ++v) {
        const int kk = ks * 32 + (v & 3) * 2 + (v >> 2) * 16 + hi * 8;
        qf[ks].u[v] = qrow[kk >> 1];
      }
  }

  const unsigned short* kBase = Kb + (size_t)bh * T * D;
  const unsigned short* vBase = VT + (size_t)bh * D * T;

  // per-thread staging coordinates (fixed across key tiles)
  const int kr  = tid >> 1;                    // K row 0..63
  const int kof = (tid & 1) * 64;              // K col offset (halves)
  const unsigned ldsK = (unsigned)(size_t)&Ks[kr * 136 + kof];
  const unsigned ldsV = (unsigned)(size_t)&Vs[tid * 72];

  float m_i[8], l_i[8];
  v8f o[8];
#pragma unroll
  for (int r = 0; r < 8; ++r) { m_i[r] = -1e30f; l_i[r] = 0.f; }
#pragma unroll
  for (int s = 0; s < 8; ++s)
#pragma unroll
    for (int r = 0; r < 8; ++r) o[s][r] = 0.f;

  for (int jt = 0; jt <= qt; ++jt) {
    const int kb = jt * 64;
    // ---- async-stage K tile [64 keys][128 d] (8 x 16B per thread) ----
    {
      const unsigned short* g = kBase + (size_t)(kb + kr) * D + kof;
      ASYNC_LDS_B128(ldsK, g, "0");
      ASYNC_LDS_B128(ldsK, g, "16");
      ASYNC_LDS_B128(ldsK, g, "32");
      ASYNC_LDS_B128(ldsK, g, "48");
      ASYNC_LDS_B128(ldsK, g, "64");
      ASYNC_LDS_B128(ldsK, g, "80");
      ASYNC_LDS_B128(ldsK, g, "96");
      ASYNC_LDS_B128(ldsK, g, "112");
    }
    // ---- async-stage V^T tile [128 d][64 keys] (8 x 16B per thread) ----
    {
      const unsigned short* g = vBase + (size_t)tid * T + kb;
      ASYNC_LDS_B128(ldsV, g, "0");
      ASYNC_LDS_B128(ldsV, g, "16");
      ASYNC_LDS_B128(ldsV, g, "32");
      ASYNC_LDS_B128(ldsV, g, "48");
      ASYNC_LDS_B128(ldsV, g, "64");
      ASYNC_LDS_B128(ldsV, g, "80");
      ASYNC_LDS_B128(ldsV, g, "96");
      ASYNC_LDS_B128(ldsV, g, "112");
    }
    // prefetch next key tile into L2/WGP$ while we compute on this one
    if (jt < qt) {
      const int kn = kb + 64;
      __builtin_prefetch(kBase + (size_t)(kn + kr) * D + kof, 0, 1);
      __builtin_prefetch(vBase + (size_t)tid * T + kn, 0, 1);
    }
    asm volatile("s_wait_asynccnt 0x0" ::: "memory");
    __syncthreads();

    // ---- S = Q K^T : 4 key sub-tiles x 4 D-steps ----
    v8f sacc[4];
#pragma unroll
    for (int s = 0; s < 4; ++s)
#pragma unroll
      for (int r = 0; r < 8; ++r) sacc[s][r] = 0.f;

#pragma unroll
    for (int s = 0; s < 4; ++s) {
      const unsigned* krow = (const unsigned*)&Ks[(s * 16 + lo) * 136];
#pragma unroll
      for (int ks = 0; ks < 4; ++ks) {
        FragBF bf_;
#pragma unroll
        for (int v = 0; v < 8; ++v) {
          const int kk = ks * 32 + (v & 3) * 2 + (v >> 2) * 16 + hi * 8;
          bf_.u[v] = krow[kk >> 1];
        }
        sacc[s] = __builtin_amdgcn_wmma_f32_16x16x32_bf16(false, qf[ks].v, false, bf_.v,
                                                          (short)0, sacc[s], false, false);
      }
    }

    // ---- scale + causal mask (diagonal tile only) ----
    const bool diag = (jt == qt);
#pragma unroll
    for (int s = 0; s < 4; ++s) {
      const int col = s * 16 + lo;
#pragma unroll
      for (int r = 0; r < 8; ++r) {
        float sv = sacc[s][r] * scale;
        if (diag && col > (wave * 16 + r + 8 * hi)) sv = -1e30f;
        sacc[s][r] = sv;
      }
    }

    // ---- online softmax ----
    float alpha[8];
#pragma unroll
    for (int r = 0; r < 8; ++r) {
      float t0 = fmaxf(fmaxf(sacc[0][r], sacc[1][r]), fmaxf(sacc[2][r], sacc[3][r]));
      t0 = rowmax16(t0);
      const float mn = fmaxf(m_i[r], t0);
      alpha[r] = __expf(m_i[r] - mn);
      m_i[r] = mn;
    }
#pragma unroll
    for (int s = 0; s < 4; ++s)
#pragma unroll
      for (int r = 0; r < 8; ++r) sacc[s][r] = __expf(sacc[s][r] - m_i[r]);
#pragma unroll
    for (int r = 0; r < 8; ++r) {
      float t = sacc[0][r] + sacc[1][r] + sacc[2][r] + sacc[3][r];
      l_i[r] = l_i[r] * alpha[r] + rowsum16(t);
    }
#pragma unroll
    for (int s = 0; s < 8; ++s)
#pragma unroll
      for (int r = 0; r < 8; ++r) o[s][r] *= alpha[r];

    // ---- P (C-layout) -> LDS -> A-layout ----
#pragma unroll
    for (int s = 0; s < 4; ++s)
#pragma unroll
      for (int r = 0; r < 8; ++r)
        Ps[(wave * 16 + r + 8 * hi) * 68 + s * 16 + lo] = f2bf(sacc[s][r]);

    // ---- O += P V : 2 key K-steps x 8 d sub-tiles ----
#pragma unroll
    for (int ks2 = 0; ks2 < 2; ++ks2) {
      FragBF pf;
      const unsigned* pr = (const unsigned*)&Ps[(wave * 16 + lo) * 68];
#pragma unroll
      for (int v = 0; v < 8; ++v) {
        const int kk = ks2 * 32 + (v & 3) * 2 + (v >> 2) * 16 + hi * 8;
        pf.u[v] = pr[kk >> 1];
      }
#pragma unroll
      for (int ds = 0; ds < 8; ++ds) {
        FragBF vf;
        const unsigned* vr = (const unsigned*)&Vs[(ds * 16 + lo) * 72];
#pragma unroll
        for (int v = 0; v < 8; ++v) {
          const int kk = ks2 * 32 + (v & 3) * 2 + (v >> 2) * 16 + hi * 8;
          vf.u[v] = vr[kk >> 1];
        }
        o[ds] = __builtin_amdgcn_wmma_f32_16x16x32_bf16(false, pf.v, false, vf.v,
                                                        (short)0, o[ds], false, false);
      }
    }
    __syncthreads();
  }

  // ---- write y (bf16, [B,T,C] with head concat) ----
#pragma unroll
  for (int r = 0; r < 8; ++r) {
    const float linv = 1.f / l_i[r];
    const int t = qb + wave * 16 + r + 8 * hi;
#pragma unroll
    for (int ds = 0; ds < 8; ++ds) {
      const int d = ds * 16 + lo;
      Y[((size_t)b * T + t) * C + h * D + d] = f2bf(o[ds][r] * linv);
    }
  }
}

// ---------------------------------------------------------------------------
extern "C" void kernel_launch(void* const* d_in, const int* in_sizes, int n_in,
                              void* d_out, int out_size, void* d_ws, size_t ws_size,
                              hipStream_t stream) {
  (void)in_sizes; (void)n_in; (void)out_size; (void)ws_size;
  const float* x      = (const float*)d_in[0];   // [4,2048,2048]
  const float* w_attn = (const float*)d_in[1];   // [2048,6144]
  const float* b_attn = (const float*)d_in[2];   // [6144]
  const float* w_proj = (const float*)d_in[3];   // [2048,2048]
  const float* b_proj = (const float*)d_in[4];   // [2048]
  float* out = (float*)d_out;                    // [4,2048,2048]

  const int B = 4, T = 2048, C = 2048, H = 16, D = 128;
  const int M = B * T;                           // 8192

  size_t off = 0;
  char* wsb = (char*)d_ws;
  auto wsAlloc = [&](size_t bytes) -> void* {
    void* p = wsb + off;
    off += (bytes + 255) & ~(size_t)255;
    return p;
  };
  unsigned short* xb  = (unsigned short*)wsAlloc((size_t)M * C * 2);
  unsigned short* wab = (unsigned short*)wsAlloc((size_t)C * 3 * C * 2);
  unsigned short* wpb = (unsigned short*)wsAlloc((size_t)C * C * 2);
  unsigned short* qbf = (unsigned short*)wsAlloc((size_t)B * H * T * D * 2);
  unsigned short* kbf = (unsigned short*)wsAlloc((size_t)B * H * T * D * 2);
  unsigned short* vtb = (unsigned short*)wsAlloc((size_t)B * H * D * T * 2);
  unsigned short* yb  = (unsigned short*)wsAlloc((size_t)M * C * 2);

  // 1) fp32 -> bf16 conversions
  {
    long long n4 = (long long)M * C / 4;
    cvt_bf16_kernel<<<(unsigned)((n4 + 255) / 256), 256, 0, stream>>>(x, xb, n4);
    n4 = (long long)C * 3 * C / 4;
    cvt_bf16_kernel<<<(unsigned)((n4 + 255) / 256), 256, 0, stream>>>(w_attn, wab, n4);
    n4 = (long long)C * C / 4;
    cvt_bf16_kernel<<<(unsigned)((n4 + 255) / 256), 256, 0, stream>>>(w_proj, wpb, n4);
  }

  // 2) QKV GEMM: [8192x2048] x [2048x6144] -> q/k/vT (bf16, head-split layouts)
  {
    QKVEpi epi{b_attn, qbf, kbf, vtb};
    gemm_bf16<QKVEpi><<<dim3(3 * C / 128, M / 128), 256, 0, stream>>>(
        xb, wab, M, 3 * C, C, epi);
  }

  // 3) flash attention -> y bf16 [B,T,C]
  attn_kernel<<<dim3(T / 64, B * H), 128, 0, stream>>>(qbf, kbf, vtb, yb);

  // 4) projection GEMM: [8192x2048] x [2048x2048] + bias -> out fp32
  {
    ProjEpi epi{b_proj, out, C};
    gemm_bf16<ProjEpi><<<dim3(C / 128, M / 128), 256, 0, stream>>>(
        yb, wpb, M, C, C, epi);
  }
}
